// ResidualVectorQuantizer_16063177687198
// MI455X (gfx1250) — matrix operand order repaired and checked by reading, not compiled, not run on hardware
//
#include <hip/hip_runtime.h>
#include <hip/hip_bf16.h>

// ---------------- problem constants ----------------
constexpr int NQ   = 4;
constexpr int NE   = 2048;
constexpr int EDIM = 128;
constexpr int NTOK = 32768;
constexpr float BETA = 0.25f;

// ---------------- d_out layout (floats) ----------------
constexpr size_t OFF_XQ   = 0;
constexpr size_t OFF_RES  = (size_t)NTOK * EDIM;                  // 4194304
constexpr size_t OFF_LOSS = OFF_RES + (size_t)NTOK * EDIM;        // 8388608
constexpr size_t OFF_IND  = OFF_LOSS + 1;                         // 8388609
constexpr size_t OFF_DIST = OFF_IND + (size_t)NTOK * NQ;          // 8519681

// ---------------- workspace layout (bytes) ----------------
constexpr size_t WS_RESID = 0;                                    // f32 [NTOK*EDIM]
constexpr size_t WS_XQACC = WS_RESID + (size_t)NTOK*EDIM*4;       // f32 [NTOK*EDIM]
constexpr size_t WS_RHI   = WS_XQACC + (size_t)NTOK*EDIM*4;       // u16 [NTOK*EDIM]
constexpr size_t WS_RLO   = WS_RHI   + (size_t)NTOK*EDIM*2;       // u16 [NTOK*EDIM]
constexpr size_t WS_RNORM = WS_RLO   + (size_t)NTOK*EDIM*2;       // f32 [NTOK]
constexpr size_t WS_EHI   = WS_RNORM + (size_t)NTOK*4;            // u16 [NQ*NE*EDIM]
constexpr size_t WS_ELO   = WS_EHI   + (size_t)NQ*NE*EDIM*2;      // u16 [NQ*NE*EDIM]
constexpr size_t WS_ENORM = WS_ELO   + (size_t)NQ*NE*EDIM*2;      // f32 [NQ*NE]
constexpr size_t WS_IDX   = WS_ENORM + (size_t)NQ*NE*4;           // i32 [NQ*NTOK]
constexpr size_t WS_LOSS  = WS_IDX   + (size_t)NQ*NTOK*4;         // f32 [NQ]

// ---------------- types ----------------
typedef __attribute__((ext_vector_type(16))) __bf16 v16bf;
typedef __attribute__((ext_vector_type(8)))  float  v8f;

union FragBF {           // 16 bf16 = 8 VGPRs, loadable as two 16B chunks
    v16bf v;
    uint4 u[2];
};

__device__ __forceinline__ unsigned short f2bf_rne(float f) {
    unsigned int u = __float_as_uint(f);
    u += 0x7FFFu + ((u >> 16) & 1u);   // round-to-nearest-even into bf16
    return (unsigned short)(u >> 16);
}
__device__ __forceinline__ float bf2f(unsigned short b) {
    return __uint_as_float(((unsigned int)b) << 16);
}

// ---------------- TDM availability probe ----------------
#if defined(__AMDGCN__) && defined(__has_builtin)
#if __has_builtin(__builtin_amdgcn_tensor_load_to_lds) && \
    __has_builtin(__builtin_amdgcn_s_wait_tensorcnt)
#define USE_TDM 1
#endif
#endif
#ifndef USE_TDM
#define USE_TDM 0
#endif

// LDS tile row pitch: 256B of data + 16B TDM pad = 272B -> 136 ushorts.
constexpr int LDSTRIDE = 136;
constexpr int TILE_USH = 64 * LDSTRIDE;                      // one 64-row array
constexpr unsigned BUF_STRIDE = 2u * TILE_USH;               // hi+lo per buffer
constexpr size_t DIST_LDS_BYTES = (size_t)4 * TILE_USH * 2;  // 2 bufs x (hi+lo)

#if USE_TDM
typedef __attribute__((ext_vector_type(4))) unsigned int u32x4;
typedef __attribute__((ext_vector_type(8))) int          i32x8;
typedef __attribute__((ext_vector_type(4))) int          i32x4;

// One TDM descriptor: copy a 64-row x 256B tile (row pitch 256B in global)
// into LDS with 16B padding after every 256B row (272B LDS pitch).
__device__ __forceinline__ void tdm_load_tile(const unsigned short* gsrc,
                                              unsigned lds_byte) {
    const unsigned long long ga = (unsigned long long)(uintptr_t)gsrc;
    u32x4 g0;
    g0.x = 1u;                                            // count=1, user desc
    g0.y = lds_byte;                                      // lds_addr[31:0]
    g0.z = (unsigned)(ga & 0xFFFFFFFFull);                // global_addr[31:0]
    g0.w = (unsigned)((ga >> 32) & 0x01FFFFFFu)           // global_addr[56:32]
         | (2u << 30);                                    // type=2 ("image")
    i32x8 g1;
    g1.s0 = (3 << 16)      // data_size = 8B elements
          | (1 << 20)      // pad_enable
          | (5 << 22)      // pad_interval: every 64 DWORDs (256B)
          | (3 << 25);     // pad_amount: 4 DWORDs (16B)
    g1.s1 = (32 << 16);    // tensor_dim0 = 32 (x8B = 256B row)
    g1.s2 = (64 << 16);    // tensor_dim1 = 64 rows
    g1.s3 = (32 << 16);    // tile_dim0 = 32
    g1.s4 = 64;            // tile_dim1 = 64 (tile_dim2 = 0 -> 2D)
    g1.s5 = 32;            // tensor_dim0_stride = 32 (8B units)
    g1.s6 = 0;
    g1.s7 = 0;
    i32x4 z4 = {};
#if __clang_major__ >= 23
    i32x8 z8 = {};
    __builtin_amdgcn_tensor_load_to_lds(g0, g1, z4, z4, z8, 0);
#else
    __builtin_amdgcn_tensor_load_to_lds(g0, g1, z4, z4, 0);
#endif
}
#endif // USE_TDM

// ============================================================
// Kernel 1: split codebook to bf16 hi/lo + per-entry squared norms
// ============================================================
__global__ __launch_bounds__(EDIM)
void prep_emb_kernel(const float* __restrict__ emb,
                     unsigned short* __restrict__ ehi,
                     unsigned short* __restrict__ elo,
                     float* __restrict__ enorm) {
    __shared__ float red[EDIM];
    const int row = blockIdx.x;
    const int t   = threadIdx.x;
    const size_t o = (size_t)row * EDIM + t;
    float e = emb[o];
    unsigned short h = f2bf_rne(e);
    ehi[o] = h;
    elo[o] = f2bf_rne(e - bf2f(h));
    red[t] = e * e;
    __syncthreads();
    #pragma unroll
    for (int s = EDIM / 2; s > 0; s >>= 1) {
        if (t < s) red[t] += red[t + s];
        __syncthreads();
    }
    if (t == 0) enorm[row] = red[0];
}

// ============================================================
// Kernel 2: init residual = x, xq_acc = 0, bf16 split, row norms
// ============================================================
__global__ __launch_bounds__(EDIM)
void init_tokens_kernel(const float* __restrict__ x,
                        float* __restrict__ residual,
                        float* __restrict__ xqacc,
                        unsigned short* __restrict__ rhi,
                        unsigned short* __restrict__ rlo,
                        float* __restrict__ rnorm) {
    __shared__ float red[EDIM];
    const int tok = blockIdx.x;
    const int t   = threadIdx.x;
    const size_t o = (size_t)tok * EDIM + t;
    float r = x[o];
    residual[o] = r;
    xqacc[o]    = 0.0f;
    unsigned short h = f2bf_rne(r);
    rhi[o] = h;
    rlo[o] = f2bf_rne(r - bf2f(h));
    red[t] = r * r;
    __syncthreads();
    #pragma unroll
    for (int s = EDIM / 2; s > 0; s >>= 1) {
        if (t < s) red[t] += red[t + s];
        __syncthreads();
    }
    if (t == 0) rnorm[tok] = red[0];
}

__global__ void zero_loss_kernel(float* __restrict__ loss) {
    if (threadIdx.x < NQ) loss[threadIdx.x] = 0.0f;
}

// ============================================================
// Kernel 3: distances (WMMA bf16 x3 split) + fused argmin
// block = 256 threads (8 waves); block tile = 128 tokens x all 2048 entries
// wave tile = 16 tokens; N-tiles of 64 entries, TDM double-buffered in LDS
// ============================================================
__device__ __forceinline__ v8f wmma_bf16(v16bf a, v16bf b, v8f c) {
    return __builtin_amdgcn_wmma_f32_16x16x32_bf16(false, a, false, b,
                                                   (short)0, c, false, false);
}

__global__ __launch_bounds__(256)
void dist_argmin_kernel(const unsigned short* __restrict__ rhi,
                        const unsigned short* __restrict__ rlo,
                        const float* __restrict__ rnorm,
                        const unsigned short* __restrict__ ehi,
                        const unsigned short* __restrict__ elo,
                        const float* __restrict__ enorm,
                        float* __restrict__ out,
                        int* __restrict__ idxws,
                        int q) {
    // ping-pong buffers addressed by integer offsets off one LDS base
    // (no pointer arrays from LDS symbols: backend can't relocate those)
    extern __shared__ unsigned short smem[];

    const int tid  = threadIdx.x;
    const int lane = tid & 31;
    const int wave = tid >> 5;
    const int half = lane >> 4;     // 0: M=j rows,   1: M=j+8 rows (C layout)
    const int l16  = lane & 15;
    const int wtok = blockIdx.x * 128 + wave * 16;

    // ---- A fragments (16 tokens x K=128, hi & lo), reused over all N-tiles ----
    // CDNA5 16-bit A 16x32 layout: lane m = l16; half 0 -> K {0..7,16..23},
    // half 1 -> {8..15,24..31}; per k-step: two contiguous 16B chunks.
    FragBF a_hi[4], a_lo[4];
    {
        const unsigned short* rh = rhi + (size_t)(wtok + l16) * EDIM;
        const unsigned short* rl = rlo + (size_t)(wtok + l16) * EDIM;
        #pragma unroll
        for (int kk = 0; kk < 4; ++kk) {
            const int k0 = kk * 32 + half * 8;
            a_hi[kk].u[0] = *reinterpret_cast<const uint4*>(rh + k0);
            a_hi[kk].u[1] = *reinterpret_cast<const uint4*>(rh + k0 + 16);
            a_lo[kk].u[0] = *reinterpret_cast<const uint4*>(rl + k0);
            a_lo[kk].u[1] = *reinterpret_cast<const uint4*>(rl + k0 + 16);
        }
    }

    float rn[8];
    #pragma unroll
    for (int j = 0; j < 8; ++j) rn[j] = rnorm[wtok + j + 8 * half];

    float minv[8];
    int   mini[8];
    #pragma unroll
    for (int j = 0; j < 8; ++j) { minv[j] = 3.4e38f; mini[j] = 0; }

    const unsigned short* eh0 = ehi + (size_t)q * NE * EDIM;
    const unsigned short* el0 = elo + (size_t)q * NE * EDIM;
    const float*          en0 = enorm + q * NE;

    constexpr int NTILES = NE / 64;

#if USE_TDM
    const unsigned smem_byte = (unsigned)(uintptr_t)smem;  // raw LDS byte addr
    if (wave == 0) {   // prime the pipeline: tile 0 -> buffer 0
        tdm_load_tile(eh0, smem_byte);
        tdm_load_tile(el0, smem_byte + (unsigned)TILE_USH * 2u);
    }
#endif

    for (int nt = 0; nt < NTILES; ++nt) {
        const unsigned cur_off = (nt & 1) ? BUF_STRIDE : 0u;  // ushort units
        const unsigned short* sh = smem + cur_off;
        const unsigned short* sl = smem + cur_off + TILE_USH;

#if USE_TDM
        if (wave == 0) {
            if (nt + 1 < NTILES) {
                const unsigned nxt_off = cur_off ^ BUF_STRIDE;
                const size_t g = (size_t)(nt + 1) * 64 * EDIM;
                tdm_load_tile(eh0 + g, smem_byte + nxt_off * 2u);
                tdm_load_tile(el0 + g,
                              smem_byte + (nxt_off + (unsigned)TILE_USH) * 2u);
                __builtin_amdgcn_s_wait_tensorcnt(2);   // current tile retired
            } else {
                __builtin_amdgcn_s_wait_tensorcnt(0);   // drain
            }
        }
        __syncthreads();
#else
        __syncthreads();
        // synchronous fallback: stage 64 entries x 128 k (hi & lo) into LDS
        unsigned short* shw = smem + cur_off;
        unsigned short* slw = smem + cur_off + TILE_USH;
        #pragma unroll
        for (int i = 0; i < 4; ++i) {
            const int c   = tid + i * 256;      // 0..1023
            const int row = c >> 4;
            const int ch  = c & 15;
            const size_t g = ((size_t)(nt * 64 + row)) * EDIM + ch * 8;
            *reinterpret_cast<uint4*>(shw + row * LDSTRIDE + ch * 8) =
                *reinterpret_cast<const uint4*>(eh0 + g);
            *reinterpret_cast<uint4*>(slw + row * LDSTRIDE + ch * 8) =
                *reinterpret_cast<const uint4*>(el0 + g);
        }
        __syncthreads();
#endif

        #pragma unroll
        for (int ns = 0; ns < 4; ++ns) {
            v8f acc = {};
            // B 32x16 layout: lane n = l16; half 0 -> K {0..7,16..23}, half 1 -> K {8..15,24..31};
            // two 16B LDS reads per lane per k-step.
            const unsigned short* bh = sh + (ns * 16 + l16) * LDSTRIDE;
            const unsigned short* bl = sl + (ns * 16 + l16) * LDSTRIDE;
            #pragma unroll
            for (int kk = 0; kk < 4; ++kk) {
                const int k0 = kk * 32 + half * 8;
                FragBF b_hi, b_lo;
                b_hi.u[0] = *reinterpret_cast<const uint4*>(bh + k0);
                b_hi.u[1] = *reinterpret_cast<const uint4*>(bh + k0 + 16);
                b_lo.u[0] = *reinterpret_cast<const uint4*>(bl + k0);
                b_lo.u[1] = *reinterpret_cast<const uint4*>(bl + k0 + 16);
                // fp32-accurate dot via bf16 split: hi*hi + hi*lo + lo*hi
                acc = wmma_bf16(a_hi[kk].v, b_hi.v, acc);
                acc = wmma_bf16(a_hi[kk].v, b_lo.v, acc);
                acc = wmma_bf16(a_lo[kk].v, b_hi.v, acc);
            }
            // epilogue: d = ||r||^2 + ||e||^2 - 2 * dot; streaming store + argmin
            const int n  = nt * 64 + ns * 16 + l16;
            const float en = en0[n];
            #pragma unroll
            for (int j = 0; j < 8; ++j) {
                const float d = rn[j] + en - 2.0f * acc[j];
                const int tok = wtok + j + 8 * half;
                __builtin_nontemporal_store(
                    d, out + OFF_DIST + (size_t)tok * (NQ * NE) + (size_t)q * NE + n);
                if (d < minv[j]) { minv[j] = d; mini[j] = n; }
            }
        }
        __syncthreads();   // all waves done reading before buffer is reissued
    }

    // argmin reduce across each 16-lane half (halves hold disjoint token rows)
    #pragma unroll
    for (int j = 0; j < 8; ++j) {
        float v = minv[j];
        int  ix = mini[j];
        #pragma unroll
        for (int m = 1; m < 16; m <<= 1) {
            const float ov = __shfl_xor(v, m, 32);
            const int   oi = __shfl_xor(ix, m, 32);
            if (ov < v || (ov == v && oi < ix)) { v = ov; ix = oi; }
        }
        if (l16 == 0) {
            const int tok = wtok + j + 8 * half;
            idxws[q * NTOK + tok] = ix;
            out[OFF_IND + (size_t)tok * NQ + q] = (float)ix;
        }
    }
}

// ============================================================
// Kernel 4: residual update (gather + STE collapse) per stage
// ============================================================
__global__ __launch_bounds__(EDIM)
void update_kernel(const float* __restrict__ emb,
                   float* __restrict__ residual,
                   float* __restrict__ xqacc,
                   unsigned short* __restrict__ rhi,
                   unsigned short* __restrict__ rlo,
                   float* __restrict__ rnorm,
                   const int* __restrict__ idxws,
                   float* __restrict__ loss,
                   float* __restrict__ out,
                   int q, int last) {
    __shared__ float red[EDIM];
    const int tok = blockIdx.x;
    const int t   = threadIdx.x;
    const int idx = idxws[q * NTOK + tok];
    const size_t o = (size_t)tok * EDIM + t;
    const float xq = emb[((size_t)q * NE + idx) * EDIM + t];
    const float r  = residual[o] - xq;     // r_new; (xq - r_pre)^2 == r_new^2
    const float xa = xqacc[o] + xq;
    residual[o] = r;
    xqacc[o]    = xa;
    unsigned short h = f2bf_rne(r);
    rhi[o] = h;
    rlo[o] = f2bf_rne(r - bf2f(h));
    red[t] = r * r;
    __syncthreads();
    #pragma unroll
    for (int s = EDIM / 2; s > 0; s >>= 1) {
        if (t < s) red[t] += red[t + s];
        __syncthreads();
    }
    if (t == 0) {
        rnorm[tok] = red[0];
        atomicAdd(&loss[q], red[0]);
    }
    if (last) {
        out[OFF_XQ + o]  = xa;
        out[OFF_RES + o] = r;
    }
}

__global__ void finalize_kernel(const float* __restrict__ loss,
                                float* __restrict__ out) {
    if (threadIdx.x == 0) {
        float s = 0.0f;
        #pragma unroll
        for (int q = 0; q < NQ; ++q) s += loss[q];
        const float scale = (1.0f + BETA) / ((float)NQ * (float)NTOK * (float)EDIM);
        out[OFF_LOSS] = s * scale;
    }
}

// ============================================================
extern "C" void kernel_launch(void* const* d_in, const int* in_sizes, int n_in,
                              void* d_out, int out_size, void* d_ws, size_t ws_size,
                              hipStream_t stream) {
    const float* x   = (const float*)d_in[0];
    const float* emb = (const float*)d_in[1];
    float* out = (float*)d_out;
    char*  ws  = (char*)d_ws;

    float*          residual = (float*)(ws + WS_RESID);
    float*          xqacc    = (float*)(ws + WS_XQACC);
    unsigned short* rhi      = (unsigned short*)(ws + WS_RHI);
    unsigned short* rlo      = (unsigned short*)(ws + WS_RLO);
    float*          rnorm    = (float*)(ws + WS_RNORM);
    unsigned short* ehi      = (unsigned short*)(ws + WS_EHI);
    unsigned short* elo      = (unsigned short*)(ws + WS_ELO);
    float*          enorm    = (float*)(ws + WS_ENORM);
    int*            idxws    = (int*)(ws + WS_IDX);
    float*          loss     = (float*)(ws + WS_LOSS);

    prep_emb_kernel<<<NQ * NE, EDIM, 0, stream>>>(emb, ehi, elo, enorm);
    init_tokens_kernel<<<NTOK, EDIM, 0, stream>>>(x, residual, xqacc, rhi, rlo, rnorm);
    zero_loss_kernel<<<1, 32, 0, stream>>>(loss);

    for (int q = 0; q < NQ; ++q) {
        dist_argmin_kernel<<<NTOK / 128, 256, DIST_LDS_BYTES, stream>>>(
            rhi, rlo, rnorm, ehi, elo, enorm, out, idxws, q);
        update_kernel<<<NTOK, EDIM, 0, stream>>>(
            emb, residual, xqacc, rhi, rlo, rnorm, idxws, loss, out,
            q, (q == NQ - 1) ? 1 : 0);
    }
    finalize_kernel<<<1, 1, 0, stream>>>(loss, out);
}
